// EnhancedEdgeEmbedding_43499428774417
// MI455X (gfx1250) — matrix operand order repaired and checked by reading, not compiled, not run on hardware
//
#include <hip/hip_runtime.h>

typedef __attribute__((ext_vector_type(16))) _Float16 v16h;
typedef __attribute__((ext_vector_type(8)))  _Float16 v8h;
typedef __attribute__((ext_vector_type(4)))  _Float16 v4h;
typedef __attribute__((ext_vector_type(8)))  float    v8f;
typedef __attribute__((ext_vector_type(4)))  float    v4f;
typedef __attribute__((ext_vector_type(4)))  unsigned int v4u;
typedef __attribute__((ext_vector_type(8)))  int      v8i;
typedef __attribute__((ext_vector_type(4)))  int      v4i;

#if defined(__has_builtin)
#if __has_builtin(__builtin_amdgcn_tensor_load_to_lds) && __has_builtin(__builtin_amdgcn_s_wait_tensorcnt)
#define HAVE_TDM 1
#endif
#if __has_builtin(__builtin_amdgcn_sched_barrier)
#define SCHED_FENCE() __builtin_amdgcn_sched_barrier(0)
#endif
#endif
#ifndef HAVE_TDM
#define HAVE_TDM 0
#endif
#ifndef SCHED_FENCE
#define SCHED_FENCE()
#endif

#define NODE_DIM 128
#define HD       128
#define K1       288   // 272 padded to multiple of 32
#define K2       256
#define K3       128
// padded column strides (halves) for the staged weight panels: +8 halves kills
// the power-of-two LDS bank aliasing (stride*2 bytes = 592/528/272, all 16B-mult)
#define KSP1     (K1+8)   // 296
#define KSP2     (K2+8)   // 264
#define KSP3     (K3+8)   // 136
#define SROW     296   // LDS activation row stride in halves (conflict-free)
#define WAVES    8
#define MROWS    16
#define TILE_E   (WAVES*MROWS)
#define PANEL    (16*KSP1)  // halves per staging buffer (max panel = layer-1)

// f16 weight scratch layout in d_ws (units: halves), column-major [out][KSP]
#define W1T_OFF  0
#define W1T_N    (256*KSP1)        // 75776
#define W2T_OFF  (W1T_N)
#define W2T_N    (128*KSP2)        // 33792
#define W3T_OFF  (W1T_N+W2T_N)
#define W3T_N    (128*KSP3)        // 17408
#define WS_HALFS (W1T_N+W2T_N+W3T_N) // 126976 halves = 248 KB

// ---------------- weight prep: f32 row-major -> f16 column-major, padded -------
__global__ __launch_bounds__(256)
void prep_weights_f16(const float* __restrict__ W1, const float* __restrict__ W2,
                      const float* __restrict__ W3, _Float16* __restrict__ ws)
{
  int t = blockIdx.x * 256 + threadIdx.x;
  if (t < W1T_N) {
    int col = t / KSP1, k = t % KSP1;
    ws[t] = (k < 272) ? (_Float16)W1[k * 256 + col] : (_Float16)0.0f;
  } else if (t < W2T_OFF + W2T_N) {
    int u = t - W2T_OFF; int col = u / KSP2, k = u % KSP2;
    ws[t] = (k < K2) ? (_Float16)W2[k * 128 + col] : (_Float16)0.0f;
  } else if (t < WS_HALFS) {
    int u = t - W3T_OFF; int col = u / KSP3, k = u % KSP3;
    ws[t] = (k < K3) ? (_Float16)W3[k * 128 + col] : (_Float16)0.0f;
  }
}

// ---------------- panel staging: TDM DMA (preferred) or cooperative copy -------
// Copies `halves` f16 (contiguous) from global `src` to LDS `dst`.
// TDM: single 1-row tile of 8-byte elements; issued by wave 0 only (EXEC ignored).
__device__ __forceinline__ void stage_panel(const _Float16* __restrict__ src,
                                            _Float16* dst, int halves,
                                            int tid, bool tdm_wave)
{
#if HAVE_TDM
  if (tdm_wave) {
    const unsigned n   = (unsigned)halves / 4u;        // # of 8-byte elements
    const unsigned lds = (unsigned)(uintptr_t)dst;     // LDS byte address (low 32b)
    const unsigned long long ga = (unsigned long long)(uintptr_t)src;
    // D# group0: count=1 | lds_addr | global_addr[56:0] | type=2
    v4u g0 = { 1u, lds, (unsigned)ga,
               ((unsigned)(ga >> 32) & 0x01FFFFFFu) | (2u << 30) };
    // D# group1: data_size=8B; tensor_dim0=n, tensor_dim1=1; tile_dim0=n,
    // tile_dim1=1; strides = n  (fields packed per ISA 8.4)
    v8i g1 = { (int)(3u << 16),                         // data_size=3 (8B)
               (int)((n & 0xFFFFu) << 16),              // tensor_dim0[15:0]
               (int)((n >> 16) | (1u << 16)),           // tdim0[31:16] | tdim1 lo=1
               (int)(n << 16),                          // tdim1 hi=0 | tile_dim0=n
               1,                                       // tile_dim1=1, tile_dim2=0
               (int)n,                                  // tensor_dim0_stride lo
               (int)((n & 0xFFFFu) << 16),              // s0 hi=0 | s1[15:0]
               (int)(n >> 16) };                        // tensor_dim1_stride[47:16]
    v4i g2 = { 1, 1, 0, 0 };                            // tensor_dim2=1, dim3=1
    v4i g3 = { 0, (int)(1u << 16), 0, 0 };              // tensor_dim4=1
#if __clang_major__ >= 23
    v8i g4 = {};
    __builtin_amdgcn_tensor_load_to_lds(g0, g1, g2, g3, g4, 0);
#else
    __builtin_amdgcn_tensor_load_to_lds(g0, g1, g2, g3, 0);
#endif
  }
#else
  const v8h* s = (const v8h*)src;
  v8h* d = (v8h*)dst;
  for (int c = tid; c < halves / 8; c += 256) d[c] = s[c];
#endif
}

__device__ __forceinline__ void stage_wait(bool tdm_wave)
{
#if HAVE_TDM
  if (tdm_wave) __builtin_amdgcn_s_wait_tensorcnt(0);
#endif
}

// ---------------- one MLP layer via WMMA ---------------------------------------
// A fragment (16x32 f16): lane holds row M=lane%16, halves[0..15]=K c0..c0+15,
// c0=(lane/16)*16 (contiguous K, matching the B fragment).
// B fragment (32x16 f16): lane holds col N=lane%16, 16 contiguous K starting at
// (lane/16)*16 -> one 32B contiguous load from the padded column-major panel.
// `bst` = base of TWO consecutive PANEL-sized LDS staging buffers; buffer
// selection is pure pointer arithmetic so address-space inference keeps ds_load.
template<int KCHUNKS, int NTILES, int KSP, bool FINAL>
__device__ __forceinline__ void mlp_layer(
    const _Float16* __restrict__ wt, const float* __restrict__ bias,
    _Float16* act, _Float16* bst,
    float* __restrict__ gout, int e0, int E, int tid, int lane)
{
  const int lo = lane & 15;
  const int hi = lane >> 4;
  const bool tdmw = (tid < 32);   // wave 0 drives the TDM

  // Hoist all A fragments to registers (frees the LDS tile for this layer's output)
  v16h a[KCHUNKS];
  {
    const _Float16* rowb = act + lo * SROW;
    const int c0 = hi * 16;
    #pragma unroll
    for (int kc = 0; kc < KCHUNKS; ++kc) {
      v8h p0 = *(const v8h*)(rowb + kc * 32 + c0);
      v8h p1 = *(const v8h*)(rowb + kc * 32 + c0 + 8);
      v16h f;
      #pragma unroll
      for (int i = 0; i < 8; ++i) { f[i] = p0[i]; f[i + 8] = p1[i]; }
      a[kc] = f;
    }
  }

  // Prologue: stage panel 0, publish to all waves.
  stage_panel(wt, bst, 16 * KSP, tid, tdmw);
  stage_wait(tdmw);
  __syncthreads();

  for (int nt = 0; nt < NTILES; ++nt) {
    const int curOff  = (nt & 1) * PANEL;
    const int nextOff = ((nt + 1) & 1) * PANEL;

    // Kick DMA for the next panel into the other buffer; overlaps with compute.
    if (nt + 1 < NTILES)
      stage_panel(wt + (nt + 1) * (16 * KSP), bst + nextOff, 16 * KSP, tid, tdmw);

    // Preload ALL B fragments first; the sched barrier keeps the scheduler from
    // sinking the ds_loads into the WMMA chain (one dscnt wait, then 9 WMMAs).
    v16h b[KCHUNKS];
    const _Float16* bcol = bst + curOff + lo * KSP + hi * 16;
    #pragma unroll
    for (int kc = 0; kc < KCHUNKS; ++kc)
      b[kc] = *(const v16h*)(bcol + kc * 32);
    SCHED_FENCE();

    v8f acc = {};
    #pragma unroll
    for (int kc = 0; kc < KCHUNKS; ++kc)
      acc = __builtin_amdgcn_wmma_f32_16x16x32_f16(
          false, a[kc], false, b[kc], (short)0, acc, false, false);

    const int col = nt * 16 + lo;
    const float bv = bias[col];
    if (FINAL) {
      #pragma unroll
      for (int r = 0; r < 8; ++r) {
        int er = e0 + r + 8 * hi;
        float v = acc[r] + bv;
        if (er < E) gout[(size_t)er * HD + col] = v;
      }
    } else {
      #pragma unroll
      for (int r = 0; r < 8; ++r) {
        float v = acc[r] + bv;
        v = v * __builtin_amdgcn_rcpf(1.0f + __expf(-v));   // SiLU
        act[(r + 8 * hi) * SROW + col] = (_Float16)v;
      }
    }

    stage_wait(tdmw);     // next panel landed in LDS
    __syncthreads();      // everyone done with current panel; next one published
  }
}

// ---------------- main edge-MLP kernel -----------------------------------------
__global__ __launch_bounds__(256)
void edge_mlp_wmma(const float* __restrict__ nf, const int* __restrict__ eidx,
                   const float* __restrict__ ea, const _Float16* __restrict__ ws,
                   const float* __restrict__ b1, const float* __restrict__ b2,
                   const float* __restrict__ b3, float* __restrict__ out, int E)
{
  __shared__ __align__(16) _Float16 act_s[WAVES][MROWS * SROW]; // per-wave tiles
  __shared__ __align__(16) _Float16 bstage[2 * PANEL];          // double-buffered B panel

  const int tid  = threadIdx.x;
  const int wave = tid >> 5, lane = tid & 31;
  const int lo = lane & 15, hi = lane >> 4;
  _Float16* act = act_s[wave];
  const int e0 = blockIdx.x * TILE_E + wave * MROWS;

  // Build x = [nf[src] | nf[dst] | edge_attr | 0-pad] as f16 in LDS (row-major)
  {
    int e = e0 + lo; if (e >= E) e = E - 1;
    int s = eidx[e];
    int d = eidx[E + e];
    const float* sf = nf + (size_t)s * NODE_DIM + hi * 64;
    const float* df = nf + (size_t)d * NODE_DIM + hi * 64;
    _Float16* row = act + lo * SROW;
    #pragma unroll
    for (int j = 0; j < 64; j += 4) {
      v4f f = *(const v4f*)(sf + j);
      v4h h = { (_Float16)f[0], (_Float16)f[1], (_Float16)f[2], (_Float16)f[3] };
      *(v4h*)(row + hi * 64 + j) = h;
    }
    #pragma unroll
    for (int j = 0; j < 64; j += 4) {
      v4f f = *(const v4f*)(df + j);
      v4h h = { (_Float16)f[0], (_Float16)f[1], (_Float16)f[2], (_Float16)f[3] };
      *(v4h*)(row + 128 + hi * 64 + j) = h;
    }
    if (hi == 0) {
      const float* ep = ea + (size_t)e * 16;
      #pragma unroll
      for (int j = 0; j < 16; j += 4) {
        v4f f = *(const v4f*)(ep + j);
        v4h h = { (_Float16)f[0], (_Float16)f[1], (_Float16)f[2], (_Float16)f[3] };
        *(v4h*)(row + 256 + j) = h;
      }
    } else {
      v4h z = { (_Float16)0.f, (_Float16)0.f, (_Float16)0.f, (_Float16)0.f };
      #pragma unroll
      for (int j = 0; j < 16; j += 4) *(v4h*)(row + 272 + j) = z;
    }
  }

  const _Float16* w1t = ws + W1T_OFF;
  const _Float16* w2t = ws + W2T_OFF;
  const _Float16* w3t = ws + W3T_OFF;

  mlp_layer<9, 16, KSP1, false>(w1t, b1, act, bstage, nullptr, e0, E, tid, lane);
  mlp_layer<8,  8, KSP2, false>(w2t, b2, act, bstage, nullptr, e0, E, tid, lane);
  mlp_layer<4,  8, KSP3, true >(w3t, b3, act, bstage, out,     e0, E, tid, lane);
}

// ---------------- coord MLP (3->128->3), scalar path ---------------------------
__global__ __launch_bounds__(256)
void coord_mlp(const float* __restrict__ coords, const int* __restrict__ eidx,
               const float* __restrict__ wc1, const float* __restrict__ bc1,
               const float* __restrict__ wc2, const float* __restrict__ bc2,
               float* __restrict__ outc, int E)
{
  int t = blockIdx.x * 256 + threadIdx.x;
  if (t >= E) return;
  int s = eidx[t], d = eidx[E + t];
  float r0 = coords[d * 3 + 0] - coords[s * 3 + 0];
  float r1 = coords[d * 3 + 1] - coords[s * 3 + 1];
  float r2 = coords[d * 3 + 2] - coords[s * 3 + 2];
  float o0 = bc2[0], o1 = bc2[1], o2 = bc2[2];
  #pragma unroll 4
  for (int k = 0; k < 128; ++k) {
    float h = fmaf(r0, wc1[k], fmaf(r1, wc1[128 + k], fmaf(r2, wc1[256 + k], bc1[k])));
    h = h * __builtin_amdgcn_rcpf(1.0f + __expf(-h));
    o0 = fmaf(h, wc2[k * 3 + 0], o0);
    o1 = fmaf(h, wc2[k * 3 + 1], o1);
    o2 = fmaf(h, wc2[k * 3 + 2], o2);
  }
  outc[(size_t)t * 3 + 0] = o0;
  outc[(size_t)t * 3 + 1] = o1;
  outc[(size_t)t * 3 + 2] = o2;
}

// ---------------- launch -------------------------------------------------------
extern "C" void kernel_launch(void* const* d_in, const int* in_sizes, int n_in,
                              void* d_out, int out_size, void* d_ws, size_t ws_size,
                              hipStream_t stream)
{
  const float* nf  = (const float*)d_in[0];
  const float* nc  = (const float*)d_in[1];
  const int*   ei  = (const int*)d_in[2];
  const float* ea  = (const float*)d_in[3];
  const float* W1  = (const float*)d_in[4];
  const float* b1  = (const float*)d_in[5];
  const float* W2  = (const float*)d_in[6];
  const float* b2  = (const float*)d_in[7];
  const float* W3  = (const float*)d_in[8];
  const float* b3  = (const float*)d_in[9];
  const float* Wc1 = (const float*)d_in[10];
  const float* bc1 = (const float*)d_in[11];
  const float* Wc2 = (const float*)d_in[12];
  const float* bc2 = (const float*)d_in[13];
  const int E = in_sizes[3] / 16;   // edge_attr is [E,16]
  _Float16* ws = (_Float16*)d_ws;

  prep_weights_f16<<<(WS_HALFS + 255) / 256, 256, 0, stream>>>(W1, W2, W3, ws);

  const int nblk = (E + TILE_E - 1) / TILE_E;
  edge_mlp_wmma<<<nblk, 256, 0, stream>>>(nf, ei, ea, ws, b1, b2, b3,
                                          (float*)d_out, E);

  float* outc = (float*)d_out + (size_t)E * HD;
  coord_mlp<<<(E + 255) / 256, 256, 0, stream>>>(nc, ei, Wc1, bc1, Wc2, bc2,
                                                 outc, E);
}
